// CapsuleNet_78383153152688
// MI455X (gfx1250) — matrix operand order repaired and checked by reading, not compiled, not run on hardware
//
#include <hip/hip_runtime.h>

typedef _Float16 v8h  __attribute__((ext_vector_type(8)));
typedef _Float16 v16h __attribute__((ext_vector_type(16)));
typedef float    v8f  __attribute__((ext_vector_type(8)));

union V16H { v16h v; v8h h[2]; };

// ---------------------------------------------------------------------------
// prep: conv1 weights (256,1,9,9) fp32 -> w1t[r=81][oc=256] fp32
// ---------------------------------------------------------------------------
__global__ void prep_w1t(const float* __restrict__ w, float* __restrict__ w1t) {
    int idx = blockIdx.x * 256 + threadIdx.x;   // 81*256
    int r = idx >> 8, oc = idx & 255;
    w1t[idx] = w[oc * 81 + r];
}

// ---------------------------------------------------------------------------
// prep: pcap weights (256,256,9,9) fp32 -> w2t[r=81][oc=256][ci=256] fp16
// ---------------------------------------------------------------------------
__global__ void prep_w2t(const float* __restrict__ w, _Float16* __restrict__ w2t) {
    int idx = blockIdx.x * 256 + threadIdx.x;   // 81*256*256 = 5,308,416
    int r   = idx >> 16;
    int rem = idx & 65535;
    int oc  = rem >> 8, ci = rem & 255;
    w2t[idx] = (_Float16)w[(oc * 256 + ci) * 81 + r];
}

// ---------------------------------------------------------------------------
// conv1 + bias + relu, output h in NHWC fp16: h[b][y][x][oc], 20x20x256
// ---------------------------------------------------------------------------
__global__ __launch_bounds__(256) void conv1_kernel(
        const float* __restrict__ inp, const float* __restrict__ w1t,
        const float* __restrict__ bias, _Float16* __restrict__ h) {
    __shared__ float in_s[9 * 28];
    const int b = blockIdx.x / 20, y = blockIdx.x % 20;
    const int t = threadIdx.x;
    if (t < 252) in_s[t] = inp[b * 784 + (y + t / 28) * 28 + (t % 28)];
    __syncthreads();
    float acc[20];
    const float cb = bias[t];
#pragma unroll
    for (int x = 0; x < 20; ++x) acc[x] = cb;
    for (int r = 0; r < 81; ++r) {
        const float wv = w1t[r * 256 + t];
        const float* row = &in_s[(r / 9) * 28 + (r % 9)];
#pragma unroll
        for (int x = 0; x < 20; ++x) acc[x] = fmaf(row[x], wv, acc[x]);
    }
    _Float16* hp = h + ((b * 20 + y) * 20) * 256 + t;
#pragma unroll
    for (int x = 0; x < 20; ++x) hp[x * 256] = (_Float16)fmaxf(acc[x], 0.f);
}

// ---------------------------------------------------------------------------
// primary-caps conv as implicit-GEMM, WMMA f16, fp32 accumulate:
//   M=256 (oc), N=9216 (b*36 + oy*6+ox), K = 81 taps x 256 ci
// Block tile 64(M) x 128(N), 8 waves in 2x4 grid, wave tile 32x32 (2x2 WMMA
// register blocking -> A/B fragments each reused twice). K-chunk 64, LDS
// double-buffered (55KB): next chunk's global loads issue before this chunk's
// WMMAs, one barrier per chunk.
// ---------------------------------------------------------------------------
__global__ __launch_bounds__(256) void pcap_gemm(
        const _Float16* __restrict__ w2t, const _Float16* __restrict__ h,
        const float* __restrict__ bias, float* __restrict__ p) {
    __shared__ __align__(16) _Float16 As[2][64 * 72];    // [buf][m][k] pad->72
    __shared__ __align__(16) _Float16 Bs[2][128 * 72];   // [buf][n][k] pad->72
    const int t    = threadIdx.x;
    const int n0   = blockIdx.x * 128;
    const int m0   = blockIdx.y * 64;
    const int wv   = t >> 5, lane = t & 31;
    const int m_base = (wv >> 2) * 32;       // 0,32
    const int n_base = (wv & 3) * 32;        // 0,32,64,96
    const int half = lane >> 4, l15 = lane & 15;

    // staging: row = (t>>3) + 32*rep, 8 halfwords per part (t&7)
    const int srow  = t >> 3;
    const int spart = t & 7;
    int hbase[4];
#pragma unroll
    for (int rep = 0; rep < 4; ++rep) {
        int n_g = n0 + srow + rep * 32;
        int bi = n_g / 36, s = n_g % 36;
        int oy = s / 6, ox = s % 6;
        hbase[rep] = ((bi * 20 + 2 * oy) * 20 + 2 * ox) * 256 + spart * 8;
    }
    const _Float16* wbase = w2t + (m0 + srow) * 256 + spart * 8;

    auto loadChunk = [&](int chunk, v8h* aR, v8h* bR) {
        const int r    = chunk >> 2;              // 4 ci-chunks of 64 per tap
        const int cc   = (chunk & 3) * 64;
        const int koff = ((r / 9) * 20 + (r % 9)) * 256 + cc;
#pragma unroll
        for (int rep = 0; rep < 2; ++rep)
            aR[rep] = *(const v8h*)&wbase[r * 65536 + rep * 32 * 256 + cc];
#pragma unroll
        for (int rep = 0; rep < 4; ++rep)
            bR[rep] = *(const v8h*)&h[hbase[rep] + koff];
    };
    auto storeChunk = [&](int buf, const v8h* aR, const v8h* bR) {
#pragma unroll
        for (int rep = 0; rep < 2; ++rep)
            *(v8h*)&As[buf][(srow + rep * 32) * 72 + spart * 8] = aR[rep];
#pragma unroll
        for (int rep = 0; rep < 4; ++rep)
            *(v8h*)&Bs[buf][(srow + rep * 32) * 72 + spart * 8] = bR[rep];
    };

    v8f acc00 = {}, acc01 = {}, acc10 = {}, acc11 = {};
    const int NCHUNK = 81 * 4;     // 324

    v8h aR[2], bR[4];
    loadChunk(0, aR, bR);
    storeChunk(0, aR, bR);
    __syncthreads();

    for (int chunk = 0; chunk < NCHUNK; ++chunk) {
        const int cur = chunk & 1;
        v8h aN[2], bN[4];
        if (chunk + 1 < NCHUNK) loadChunk(chunk + 1, aN, bN);  // in-flight VMEM

        const _Float16* Ab = &As[cur][0];
        const _Float16* Bb = &Bs[cur][0];
#pragma unroll
        for (int kk = 0; kk < 64; kk += 32) {
            V16H a0, a1, b0, b1;
            a0.h[0] = *(const v8h*)&Ab[(m_base + l15) * 72 + kk + half * 8];
            a0.h[1] = *(const v8h*)&Ab[(m_base + l15) * 72 + kk + 16 + half * 8];
            a1.h[0] = *(const v8h*)&Ab[(m_base + 16 + l15) * 72 + kk + half * 8];
            a1.h[1] = *(const v8h*)&Ab[(m_base + 16 + l15) * 72 + kk + 16 + half * 8];
            b0.h[0] = *(const v8h*)&Bb[(n_base + l15) * 72 + kk + half * 8];
            b0.h[1] = *(const v8h*)&Bb[(n_base + l15) * 72 + kk + 16 + half * 8];
            b1.h[0] = *(const v8h*)&Bb[(n_base + 16 + l15) * 72 + kk + half * 8];
            b1.h[1] = *(const v8h*)&Bb[(n_base + 16 + l15) * 72 + kk + 16 + half * 8];
            acc00 = __builtin_amdgcn_wmma_f32_16x16x32_f16(
                false, a0.v, false, b0.v, (short)0, acc00, false, false);
            acc01 = __builtin_amdgcn_wmma_f32_16x16x32_f16(
                false, a0.v, false, b1.v, (short)0, acc01, false, false);
            acc10 = __builtin_amdgcn_wmma_f32_16x16x32_f16(
                false, a1.v, false, b0.v, (short)0, acc10, false, false);
            acc11 = __builtin_amdgcn_wmma_f32_16x16x32_f16(
                false, a1.v, false, b1.v, (short)0, acc11, false, false);
        }
        if (chunk + 1 < NCHUNK) storeChunk(cur ^ 1, aN, bN);
        __syncthreads();
    }

    // epilogue: lane owns column n_g, rows half*8..half*8+7 of each subtile
#pragma unroll
    for (int sm = 0; sm < 2; ++sm) {
#pragma unroll
        for (int sn = 0; sn < 2; ++sn) {
            const int oc0 = m0 + m_base + sm * 16 + half * 8;
            const int n_g = n0 + n_base + sn * 16 + l15;
            v8f a = sm ? (sn ? acc11 : acc10) : (sn ? acc01 : acc00);
            float* dst = p + n_g * 256 + oc0;
#pragma unroll
            for (int i = 0; i < 8; ++i) dst[i] = a[i] + bias[oc0 + i];
        }
    }
}

// ---------------------------------------------------------------------------
// squash primary capsules: p[b][s][n32*8+d] -> caps[b][n][d],
// n = n32*36 + ox*6 + oy  (matches reference swapaxes(2,4))
// ---------------------------------------------------------------------------
__global__ void squash_primary(const float* __restrict__ p, float* __restrict__ caps) {
    const int idx = blockIdx.x * 256 + threadIdx.x;  // 256*36*32
    const int n32 = idx & 31;
    const int s   = (idx >> 5) % 36;
    const int b   = idx / (36 * 32);
    const float* src = p + (b * 36 + s) * 256 + n32 * 8;
    float v[8]; float ns = 0.f;
#pragma unroll
    for (int d = 0; d < 8; ++d) { v[d] = src[d]; ns += v[d] * v[d]; }
    const float scale = ns / ((1.f + ns) * sqrtf(ns + 1e-6f));
    const int oy = s / 6, ox = s % 6;
    const int n = n32 * 36 + ox * 6 + oy;
    float* dst = caps + (b * 1152 + n) * 8;
#pragma unroll
    for (int d = 0; d < 8; ++d) dst[d] = v[d] * scale;
}

// ---------------------------------------------------------------------------
// fused u_hat + 3 dynamic-routing iterations per (b,j). u_hat kept in LDS
// (fp16, 36KB) -> 188MB intermediate never touches HBM.
// ---------------------------------------------------------------------------
__global__ __launch_bounds__(256) void routing_kernel(
        const float* __restrict__ caps, const float* __restrict__ W,
        float* __restrict__ out) {
    __shared__ _Float16 uh[1152 * 16];
    __shared__ float blog[1152];
    __shared__ float cex[1152];
    __shared__ float red[256];
    __shared__ float ssh[16];
    __shared__ float vsh[16];
    const int t = threadIdx.x;
    const int j = blockIdx.x % 10;
    const int b = blockIdx.x / 10;
    const float* cb = caps + b * 1152 * 8;
    const float* Wj = W + j * 1152 * 128;

    for (int idx = t; idx < 1152 * 16; idx += 256) {
        const int n = idx >> 4, k = idx & 15;
        const float* cr = cb + n * 8;
        const float* wr = Wj + n * 128 + k;
        float sum = 0.f;
#pragma unroll
        for (int d = 0; d < 8; ++d) sum = fmaf(cr[d], wr[d * 16], sum);
        uh[idx] = (_Float16)sum;
    }
    for (int n = t; n < 1152; n += 256) blog[n] = 0.f;
    __syncthreads();

    for (int it = 0; it < 3; ++it) {
        // softmax over n
        float m_t = -1e30f;
        for (int n = t; n < 1152; n += 256) m_t = fmaxf(m_t, blog[n]);
        red[t] = m_t; __syncthreads();
        for (int s = 128; s > 0; s >>= 1) {
            if (t < s) red[t] = fmaxf(red[t], red[t + s]);
            __syncthreads();
        }
        const float mx = red[0]; __syncthreads();
        float s_t = 0.f;
        for (int n = t; n < 1152; n += 256) {
            float e = __expf(blog[n] - mx); cex[n] = e; s_t += e;
        }
        red[t] = s_t; __syncthreads();
        for (int s = 128; s > 0; s >>= 1) {
            if (t < s) red[t] += red[t + s];
            __syncthreads();
        }
        const float inv = 1.f / red[0]; __syncthreads();
        for (int n = t; n < 1152; n += 256) cex[n] *= inv;
        __syncthreads();
        // s[k] = sum_n c[n] * u_hat[n,k]  (16 k-lanes x 16 n-chunks)
        const int k = t & 15, ch = t >> 4;
        float part = 0.f;
        for (int n = ch * 72; n < ch * 72 + 72; ++n)
            part = fmaf(cex[n], (float)uh[n * 16 + k], part);
        red[t] = part; __syncthreads();
        if (t < 16) {
            float sv = 0.f;
#pragma unroll
            for (int c2 = 0; c2 < 16; ++c2) sv += red[c2 * 16 + t];
            ssh[t] = sv;
        }
        __syncthreads();
        if (t < 16) {
            float ns = 0.f;
#pragma unroll
            for (int kk = 0; kk < 16; ++kk) ns += ssh[kk] * ssh[kk];
            const float scale = ns / ((1.f + ns) * sqrtf(ns + 1e-6f));
            vsh[t] = ssh[t] * scale;
        }
        __syncthreads();
        if (it < 2) {
            for (int n = t; n < 1152; n += 256) {
                float dot = 0.f;
#pragma unroll
                for (int kk = 0; kk < 16; ++kk)
                    dot = fmaf((float)uh[n * 16 + kk], vsh[kk], dot);
                blog[n] += dot;
            }
            __syncthreads();
        }
    }
    if (t < 16) out[(b * 10 + j) * 16 + t] = vsh[t];
}

// ---------------------------------------------------------------------------
extern "C" void kernel_launch(void* const* d_in, const int* in_sizes, int n_in,
                              void* d_out, int out_size, void* d_ws, size_t ws_size,
                              hipStream_t stream) {
    const float* inp     = (const float*)d_in[0];
    const float* conv1_w = (const float*)d_in[1];
    const float* conv1_b = (const float*)d_in[2];
    const float* pcap_w  = (const float*)d_in[3];
    const float* pcap_b  = (const float*)d_in[4];
    const float* Wmat    = (const float*)d_in[5];
    float* out = (float*)d_out;

    char* ws = (char*)d_ws;
    size_t off = 0;
    auto carve = [&](size_t bytes) -> void* {
        void* ptr = ws + off;
        off = (off + bytes + 255) & ~(size_t)255;
        return ptr;
    };
    float*    w1t  = (float*)   carve((size_t)81 * 256 * 4);             // 83 KB
    _Float16* w2t  = (_Float16*)carve((size_t)81 * 256 * 256 * 2);       // 10.6 MB
    _Float16* h    = (_Float16*)carve((size_t)256 * 20 * 20 * 256 * 2);  // 52.4 MB
    float*    p    = (float*)   carve((size_t)256 * 36 * 256 * 4);       // 9.4 MB
    float*    caps = (float*)   carve((size_t)256 * 1152 * 8 * 4);       // 9.4 MB

    prep_w1t<<<81, 256, 0, stream>>>(conv1_w, w1t);
    prep_w2t<<<20736, 256, 0, stream>>>(pcap_w, w2t);
    conv1_kernel<<<256 * 20, 256, 0, stream>>>(inp, w1t, conv1_b, h);
    pcap_gemm<<<dim3(9216 / 128, 256 / 64), 256, 0, stream>>>(w2t, h, pcap_b, p);
    squash_primary<<<(256 * 36 * 32) / 256, 256, 0, stream>>>(p, caps);
    routing_kernel<<<256 * 10, 256, 0, stream>>>(caps, Wmat, out);
}
